// SSD_40269613367816
// MI455X (gfx1250) — compile-verified, matrix operand order checked by reading, not run on hardware
//
#include <hip/hip_runtime.h>
#include <hip/hip_bf16.h>

// ---------------------------------------------------------------------------
// Types for CDNA5 WMMA (wave32, 16x16x32 bf16 -> f32)
// ---------------------------------------------------------------------------
typedef __bf16 v16bf __attribute__((ext_vector_type(16)));
typedef __bf16 v8bf  __attribute__((ext_vector_type(8)));
typedef float  v8f   __attribute__((ext_vector_type(8)));

#define COUT_TILE 64
#define PIX_TILE  128
#define LDS_DW_STRIDE 20   // 16 dwords payload + 4 pad (80B row, 16B aligned)

// ---------------------------------------------------------------------------
// Weight pack: OIHW f32 -> O,(kh,kw,ciPadded) bf16 scaled by BN scale; fold bias.
// ---------------------------------------------------------------------------
__global__ void ssd_pack_weights(const float* __restrict__ w,
                                 const float* __restrict__ b,
                                 const float* __restrict__ s,
                                 const float* __restrict__ t,
                                 __bf16* __restrict__ wout,
                                 float* __restrict__ bout,
                                 int Co, int Ci, int CiP, int K)
{
    size_t total = (size_t)Co * K * K * CiP;
    size_t i = (size_t)blockIdx.x * blockDim.x + threadIdx.x;
    if (i < total) {
        int ci = (int)(i % CiP);
        size_t r = i / CiP;
        int kw = (int)(r % K); r /= K;
        int kh = (int)(r % K);
        int co = (int)(r / K);
        float val = 0.0f;
        if (ci < Ci) {
            float sc = s ? s[co] : 1.0f;
            val = w[(((size_t)co * Ci + ci) * K + kh) * K + kw] * sc;
        }
        wout[i] = (__bf16)val;
    }
    if (i < (size_t)Co) {
        float sc = s ? s[i] : 1.0f;
        float sh = t ? t[i] : 0.0f;
        bout[i] = b[i] * sc + sh;
    }
}

// ---------------------------------------------------------------------------
// NCHW f32 -> NHWC bf16 with channel zero-padding to Cp
// ---------------------------------------------------------------------------
__global__ void ssd_to_nhwc(const float* __restrict__ x, __bf16* __restrict__ y,
                            int B, int C, int Cp, int H, int W)
{
    size_t total = (size_t)B * H * W * Cp;
    size_t i = (size_t)blockIdx.x * blockDim.x + threadIdx.x;
    if (i >= total) return;
    int c = (int)(i % Cp);
    size_t r = i / Cp;
    int wx = (int)(r % W); r /= W;
    int h  = (int)(r % H);
    int bb = (int)(r / H);
    y[i] = (c < C) ? (__bf16)x[(((size_t)bb * C + c) * H + h) * W + wx]
                   : (__bf16)0.0f;
}

// ---------------------------------------------------------------------------
// 2x2 max pool, NHWC bf16 (H,W even throughout this net)
// ---------------------------------------------------------------------------
__global__ void ssd_maxpool2(const __bf16* __restrict__ in, __bf16* __restrict__ out,
                             int B, int C, int H, int W)
{
    int Ho = H >> 1, Wo = W >> 1;
    size_t total = (size_t)B * Ho * Wo * C;
    size_t i = (size_t)blockIdx.x * blockDim.x + threadIdx.x;
    if (i >= total) return;
    int c = (int)(i % C);
    size_t r = i / C;
    int ox = (int)(r % Wo); r /= Wo;
    int oy = (int)(r % Ho);
    int bb = (int)(r / Ho);
    const __bf16* p = in + (((size_t)bb * H + oy * 2) * W + ox * 2) * C + c;
    size_t rows = (size_t)W * C;
    float v0 = (float)p[0], v1 = (float)p[C];
    float v2 = (float)p[rows], v3 = (float)p[rows + C];
    float m = fmaxf(fmaxf(v0, v1), fmaxf(v2, v3));
    out[i] = (__bf16)m;
}

// ---------------------------------------------------------------------------
// Fragment load: row stored K-permuted (swap bits 3<->4 of K) so the 16 halves
// a lane needs are contiguous at column g*16 -> two 16B ds loads.
// ---------------------------------------------------------------------------
__device__ __forceinline__ v16bf ld_frag(const unsigned int* rowPtr, int g)
{
    const __bf16* p = (const __bf16*)rowPtr + g * 16;
    v8bf lo = *(const v8bf*)p;
    v8bf hi = *(const v8bf*)(p + 8);
    return __builtin_shufflevector(lo, hi,
        0,1,2,3,4,5,6,7,8,9,10,11,12,13,14,15);
}

// ---------------------------------------------------------------------------
// Implicit-GEMM conv, NHWC bf16 in/out, f32 accumulate via WMMA.
//
// Block = 256 threads = 8 waves; block tile = 64 cout x 128 pixels.
// Wave (mw = w&1, nw = w>>1) computes a 32x32 sub-tile: 2 A-frags x 2 B-frags
// -> 4 accumulators, 4 WMMAs per K-chunk (max fragment reuse).
// Ci pre-padded to mult-of-32 -> all global/LDS accesses 16B aligned.
//
// Software pipeline with double-buffered LDS: one barrier per chunk; chunk t+1
// is fetched into registers while WMMAs consume chunk t from LDS.
// Staging per thread: A one uint4 (row tid>>2, dword block tid&3 -> permuted
// base 0/8/4/12); B two uint4 (row tid>>1, halves (tid&1)*16 -> bases 0/8 or
// 4/12).
// ---------------------------------------------------------------------------
template <int K, int ST, int PD, int DL>
__global__ __launch_bounds__(256)
void ssd_conv_wmma(const __bf16* __restrict__ in, const __bf16* __restrict__ wgt,
                   const float* __restrict__ bias, __bf16* __restrict__ out,
                   int B, int H, int W, int Ci /*padded*/, int Co, int Ho, int Wo,
                   int relu)
{
    __shared__ unsigned int AsD[2][COUT_TILE][LDS_DW_STRIDE];
    __shared__ unsigned int BsD[2][PIX_TILE][LDS_DW_STRIDE];

    const int tid  = threadIdx.x;
    const int wave = tid >> 5;
    const int lane = tid & 31;
    const int mw  = wave & 1;    // 32-cout slice
    const int nw  = wave >> 1;   // 32-pixel slice
    const int g   = lane >> 4;
    const int l16 = lane & 15;

    const int pixBase = blockIdx.x * PIX_TILE;
    const int coBase  = blockIdx.y * COUT_TILE;
    const int HoWo = Ho * Wo;
    const int npix = B * HoWo;

    // A staging: row tid>>2 (0..63), dword block qa = tid&3
    const int qa   = tid & 3;
    const int arow = tid >> 2;
    const int pdA  = ((qa & 1) << 3) | ((qa & 2) << 1);
    const int aco  = coBase + arow;
    const bool avalid = aco < Co;
    const __bf16* abase = wgt + (size_t)aco * (K * K) * Ci;

    // B staging: row tid>>1 (0..127), half-block q2 = tid&1
    const int q2   = tid & 1;
    const int brow = tid >> 1;
    const int pdB  = q2 ? 4 : 0;
    const int bpix = pixBase + brow;
    const bool bvalid = bpix < npix;
    int bb = 0, oy = 0, ox = 0;
    if (bvalid) {
        bb = bpix / HoWo;
        int r = bpix - bb * HoWo;
        oy = r / Wo;
        ox = r - oy * Wo;
    }

    const int NCI = Ci >> 5;
    const int NC  = K * K * NCI;

    // loader state: coords of the NEXT chunk to fetch (division-free advance)
    int lkh = 0, lkw = 0, lcc = 0;

    auto fetch = [&](uint4& av, uint4& bv0, uint4& bv1) {
        av  = make_uint4(0u, 0u, 0u, 0u);
        bv0 = av; bv1 = av;
        if (avalid)
            av = *(const uint4*)(abase + (size_t)(lkh * K + lkw) * Ci + lcc + qa * 8);
        const int iy = oy * ST - PD + lkh * DL;
        const int ix = ox * ST - PD + lkw * DL;
        if (bvalid && iy >= 0 && iy < H && ix >= 0 && ix < W) {
            const __bf16* bp = in + (((size_t)bb * H + iy) * W + ix) * Ci + lcc + q2 * 16;
            bv0 = *(const uint4*)bp;
            bv1 = *(const uint4*)(bp + 8);
        }
        lcc += 32;
        if (lcc == Ci) { lcc = 0; ++lkw; if (lkw == K) { lkw = 0; ++lkh; } }
    };
    auto stage = [&](int buf, const uint4& av, const uint4& bv0, const uint4& bv1) {
        *(uint4*)&AsD[buf][arow][pdA]     = av;
        *(uint4*)&BsD[buf][brow][pdB]     = bv0;
        *(uint4*)&BsD[buf][brow][pdB + 8] = bv1;
    };

    v8f acc00 = {}, acc01 = {}, acc10 = {}, acc11 = {};

    uint4 av, bv0, bv1;
    fetch(av, bv0, bv1);
    stage(0, av, bv0, bv1);

    for (int t = 0; t < NC; ++t) {
        __syncthreads();                       // buf[t&1] fully staged
        const int cur = t & 1;
        const bool more = (t + 1) < NC;
        if (more) fetch(av, bv0, bv1);         // overlap with WMMAs below

        v16bf a0 = ld_frag(AsD[cur][mw * 32 + l16],       g);
        v16bf a1 = ld_frag(AsD[cur][mw * 32 + 16 + l16],  g);
        v16bf b0 = ld_frag(BsD[cur][nw * 32 + l16],       g);
        v16bf b1 = ld_frag(BsD[cur][nw * 32 + 16 + l16],  g);

        acc00 = __builtin_amdgcn_wmma_f32_16x16x32_bf16(
                    false, a0, false, b0, (short)0, acc00, false, false);
        acc01 = __builtin_amdgcn_wmma_f32_16x16x32_bf16(
                    false, a0, false, b1, (short)0, acc01, false, false);
        acc10 = __builtin_amdgcn_wmma_f32_16x16x32_bf16(
                    false, a1, false, b0, (short)0, acc10, false, false);
        acc11 = __builtin_amdgcn_wmma_f32_16x16x32_bf16(
                    false, a1, false, b1, (short)0, acc11, false, false);

        if (more) stage(cur ^ 1, av, bv0, bv1);  // other buffer: race-free
    }

    // ---- epilogue: C layout lane L, vgpr r -> M = r + 8*(L>=16), N = L&15 ----
    #pragma unroll
    for (int j = 0; j < 2; ++j) {                      // B sub-tile
        const int opix = pixBase + nw * 32 + j * 16 + l16;
        if (opix < npix) {
            int b2 = opix / HoWo;
            int r  = opix - b2 * HoWo;
            int oy2 = r / Wo;
            int ox2 = r - oy2 * Wo;
            size_t base = (((size_t)b2 * Ho + oy2) * Wo + ox2) * Co;
            #pragma unroll
            for (int i = 0; i < 2; ++i) {              // A sub-tile
                #pragma unroll
                for (int rr = 0; rr < 8; ++rr) {
                    int co = coBase + mw * 32 + i * 16 + g * 8 + rr;
                    if (co < Co) {
                        float v;
                        if (i == 0) v = (j == 0 ? acc00[rr] : acc01[rr]);
                        else        v = (j == 0 ? acc10[rr] : acc11[rr]);
                        v += bias[co];
                        if (relu) v = v > 0.0f ? v : 0.0f;
                        out[base + co] = (__bf16)v;
                    }
                }
            }
        }
    }
}

// ---------------------------------------------------------------------------
// Prior-box decode + 2-way softmax -> [B,184320,6] f32
// ---------------------------------------------------------------------------
__global__ void ssd_decode(const __bf16* __restrict__ loc0, const __bf16* __restrict__ loc1,
                           const __bf16* __restrict__ loc2, const __bf16* __restrict__ loc3,
                           const __bf16* __restrict__ cnf0, const __bf16* __restrict__ cnf1,
                           const __bf16* __restrict__ cnf2, const __bf16* __restrict__ cnf3,
                           const float* __restrict__ prior, float* __restrict__ out,
                           int total /* = B*184320 */)
{
    int i = blockIdx.x * blockDim.x + threadIdx.x;
    if (i >= total) return;
    const int NANCH = 184320;
    int b = i / NANCH;
    int g = i - b * NANCH;

    int head, local;
    if (g < 129024)      { head = 0; local = g; }
    else if (g < 170496) { head = 1; local = g - 129024; }
    else if (g < 178560) { head = 2; local = g - 170496; }
    else                 { head = 3; local = g - 178560; }

    const int fs[4] = {96, 48, 24, 24};
    const int As[4] = {14, 18, 14, 10};
    const __bf16* lp[4] = {loc0, loc1, loc2, loc3};
    const __bf16* cp[4] = {cnf0, cnf1, cnf2, cnf3};

    int A = As[head], f = fs[head];
    int a = local % A;
    int cell = local / A;                 // y*f + x (row-major)

    size_t lbase = ((size_t)b * f * f + cell) * (4 * A) + a * 4;
    size_t cbase = ((size_t)b * f * f + cell) * (2 * A) + a * 2;

    float l0 = (float)lp[head][lbase + 0];
    float l1 = (float)lp[head][lbase + 1];
    float l2 = (float)lp[head][lbase + 2];
    float l3 = (float)lp[head][lbase + 3];
    float c0 = (float)cp[head][cbase + 0];
    float c1 = (float)cp[head][cbase + 1];

    const float* pr = prior + (size_t)g * 4;
    float cx = pr[0] + l0 * 0.1f * pr[2];
    float cy = pr[1] + l1 * 0.1f * pr[3];
    float w  = pr[2] * __expf(l2 * 0.2f);
    float h  = pr[3] * __expf(l3 * 0.2f);

    float m = fmaxf(c0, c1);
    float e0 = __expf(c0 - m), e1 = __expf(c1 - m);
    float inv = 1.0f / (e0 + e1);

    float* o = out + (size_t)i * 6;
    o[0] = cx - 0.5f * w;
    o[1] = cy - 0.5f * h;
    o[2] = cx + 0.5f * w;
    o[3] = cy + 0.5f * h;
    o[4] = e0 * inv;
    o[5] = e1 * inv;
}

// ---------------------------------------------------------------------------
// Host orchestration
// ---------------------------------------------------------------------------
struct Spec { int ci, co, k, st, p, d, bn, act; };

extern "C" void kernel_launch(void* const* d_in, const int* in_sizes, int n_in,
                              void* d_out, int out_size, void* d_ws, size_t ws_size,
                              hipStream_t stream)
{
    (void)in_sizes; (void)n_in; (void)out_size; (void)ws_size;

    // ---- build the 59 conv-layer spec table (backbone 19, loc 4x5, conf 4x5) ----
    static const Spec BSPEC[19] = {
        {3,64,3,1,1,1,1,1},{64,64,3,1,1,1,1,1},
        {64,128,3,1,1,1,1,1},{128,128,3,1,1,1,1,1},
        {128,256,3,1,1,1,1,1},{256,256,3,1,1,1,1,1},{256,256,3,1,1,1,1,1},
        {256,512,3,1,1,1,1,1},{512,512,3,1,1,1,1,1},{512,512,3,1,1,1,1,1},
        {512,512,3,1,1,1,1,1},{512,512,3,1,1,1,1,1},{512,512,3,1,1,1,1,1},
        {512,512,3,1,3,3,1,1},{512,512,1,1,0,1,1,1},
        {512,256,1,1,0,1,1,1},{256,512,3,2,1,1,1,1},
        {512,256,1,1,0,1,1,1},{256,256,3,1,1,1,1,1}
    };
    Spec specs[59];
    for (int i = 0; i < 19; ++i) specs[i] = BSPEC[i];
    const int C4[4] = {512, 512, 512, 256};
    const int A4[4] = {14, 18, 14, 10};
    int n = 19;
    for (int i = 0; i < 4; ++i) {          // loc heads
        int c = C4[i], a = A4[i];
        Spec L[5] = {
            {c, c, 3, 1, 3, 3, 1, 1}, {c, c, 1, 1, 0, 1, 1, 1},
            {c, c, 3, 1, 0, 1, 0, 0}, {c, c / 2, 1, 1, 1, 1, 0, 0},
            {c / 2, 4 * a, 3, 1, 1, 1, 0, 0}
        };
        for (int j = 0; j < 5; ++j) specs[n++] = L[j];
    }
    for (int i = 0; i < 4; ++i) {          // conf heads
        int c = C4[i], a = A4[i];
        Spec L[5] = {
            {c, c, 3, 1, 3, 3, 1, 1}, {c, c, 1, 1, 0, 1, 1, 1},
            {c, c, 1, 1, 0, 1, 0, 0}, {c, c / 2, 3, 1, 1, 1, 0, 0},
            {c / 2, 2 * a, 3, 1, 1, 1, 0, 0}
        };
        for (int j = 0; j < 5; ++j) specs[n++] = L[j];
    }

    // ---- bump allocator over workspace ----
    size_t off = 0;
    auto alloc = [&](size_t bytes) -> void* {
        void* p = (char*)d_ws + off;
        off += (bytes + 255) & ~(size_t)255;
        return p;
    };

    // ---- unpack params (insertion order: x; per-layer w,b[,s,t]; prior last),
    //      pack weights (Ci padded to mult-of-32) + fold BN/bias on device ----
    const float* x = (const float*)d_in[0];
    int idx = 1;
    __bf16* wp[59];
    float*  bp[59];
    int     cip[59];
    for (int L = 0; L < 59; ++L) {
        const Spec& s = specs[L];
        const float* w  = (const float*)d_in[idx++];
        const float* b  = (const float*)d_in[idx++];
        const float* sc = nullptr;
        const float* sh = nullptr;
        if (s.bn) { sc = (const float*)d_in[idx++]; sh = (const float*)d_in[idx++]; }
        int CiP = (s.ci + 31) & ~31;
        cip[L] = CiP;
        size_t wn = (size_t)s.co * s.k * s.k * CiP;
        wp[L] = (__bf16*)alloc(wn * sizeof(__bf16));
        bp[L] = (float*)alloc((size_t)s.co * sizeof(float));
        int blocks = (int)((wn + 255) / 256);
        ssd_pack_weights<<<blocks, 256, 0, stream>>>(w, b, sc, sh, wp[L], bp[L],
                                                     s.co, s.ci, CiP, s.k);
    }
    const float* prior = (const float*)d_in[idx];

    // ---- activation / feature buffers ----
    const int Bn = 2;
    size_t PBYTES = (size_t)Bn * 768 * 768 * 64 * sizeof(__bf16);  // largest activation
    __bf16* P0 = (__bf16*)alloc(PBYTES);
    __bf16* P1 = (__bf16*)alloc(PBYTES);
    const int featF[4] = {96, 48, 24, 24};
    const int featC[4] = {512, 512, 512, 256};
    __bf16* F[4];
    for (int i = 0; i < 4; ++i)
        F[i] = (__bf16*)alloc((size_t)Bn * featF[i] * featF[i] * featC[i] * sizeof(__bf16));
    __bf16* LOC[4];
    __bf16* CONF[4];
    for (int i = 0; i < 4; ++i) {
        LOC[i]  = (__bf16*)alloc((size_t)Bn * featF[i] * featF[i] * 4 * A4[i] * sizeof(__bf16));
        CONF[i] = (__bf16*)alloc((size_t)Bn * featF[i] * featF[i] * 2 * A4[i] * sizeof(__bf16));
    }

    auto runConv = [&](int L, const __bf16* src, __bf16* dst, int H, int W,
                       int& Ho, int& Wo) {
        const Spec& s = specs[L];
        Ho = (H + 2 * s.p - s.d * (s.k - 1) - 1) / s.st + 1;
        Wo = (W + 2 * s.p - s.d * (s.k - 1) - 1) / s.st + 1;
        int npix = Bn * Ho * Wo;
        dim3 grid((npix + PIX_TILE - 1) / PIX_TILE, (s.co + COUT_TILE - 1) / COUT_TILE);
        int CiP = cip[L];
        #define CONV_ARGS src, wp[L], bp[L], dst, Bn, H, W, CiP, s.co, Ho, Wo, s.act
        if      (s.k == 3 && s.st == 1 && s.p == 1 && s.d == 1)
            ssd_conv_wmma<3,1,1,1><<<grid, 256, 0, stream>>>(CONV_ARGS);
        else if (s.k == 3 && s.st == 1 && s.p == 3 && s.d == 3)
            ssd_conv_wmma<3,1,3,3><<<grid, 256, 0, stream>>>(CONV_ARGS);
        else if (s.k == 1 && s.st == 1 && s.p == 0)
            ssd_conv_wmma<1,1,0,1><<<grid, 256, 0, stream>>>(CONV_ARGS);
        else if (s.k == 3 && s.st == 2)
            ssd_conv_wmma<3,2,1,1><<<grid, 256, 0, stream>>>(CONV_ARGS);
        else if (s.k == 3 && s.st == 1 && s.p == 0)
            ssd_conv_wmma<3,1,0,1><<<grid, 256, 0, stream>>>(CONV_ARGS);
        else  // k==1, p==1
            ssd_conv_wmma<1,1,1,1><<<grid, 256, 0, stream>>>(CONV_ARGS);
        #undef CONV_ARGS
    };

    // ---- input layout convert (3 -> 32 padded channels, NHWC bf16) ----
    {
        size_t tot = (size_t)Bn * 768 * 768 * 32;
        ssd_to_nhwc<<<(int)((tot + 255) / 256), 256, 0, stream>>>(x, P0, Bn, 3, 32, 768, 768);
    }

    // ---- backbone: 23 ops (convIdx, featTap) ; convIdx == -1 -> 2x2 maxpool ----
    struct Op { int conv; int feat; };
    static const Op ops[23] = {
        {0,-1},{1,-1},{-1,-1},{2,-1},{3,-1},{-1,-1},{4,-1},{5,-1},{6,-1},
        {-1,-1},{7,-1},{8,-1},{9,0},{-1,-1},{10,-1},{11,-1},{12,1},
        {13,-1},{14,-1},{15,-1},{16,2},{17,-1},{18,3}
    };
    __bf16* cur = P0;
    int H = 768, W = 768, C = 32;   // C=32: padded input channels
    for (int o = 0; o < 23; ++o) {
        if (ops[o].conv < 0) {
            int Ho = H >> 1, Wo = W >> 1;
            __bf16* dst = (cur == P0) ? P1 : P0;
            size_t tot = (size_t)Bn * Ho * Wo * C;
            ssd_maxpool2<<<(int)((tot + 255) / 256), 256, 0, stream>>>(cur, dst, Bn, C, H, W);
            cur = dst; H = Ho; W = Wo;
        } else {
            const Spec& s = specs[ops[o].conv];
            __bf16* dst = (ops[o].feat >= 0) ? F[ops[o].feat]
                                             : ((cur == P0) ? P1 : P0);
            int Ho, Wo;
            runConv(ops[o].conv, cur, dst, H, W, Ho, Wo);
            cur = dst; H = Ho; W = Wo; C = s.co;
        }
    }

    // ---- heads: loc chains (layers 19+i*5+j), conf chains (39+i*5+j) ----
    for (int i = 0; i < 4; ++i) {
        for (int branch = 0; branch < 2; ++branch) {
            int base = (branch == 0 ? 19 : 39) + i * 5;
            __bf16* dstFinal = (branch == 0) ? LOC[i] : CONF[i];
            const __bf16* src = F[i];
            int h = featF[i], w = featF[i];
            __bf16* ping = P0;
            for (int j = 0; j < 5; ++j) {
                __bf16* dst = (j == 4) ? dstFinal : ping;
                int Ho, Wo;
                runConv(base + j, (const __bf16*)src, dst, h, w, Ho, Wo);
                src = dst; h = Ho; w = Wo;
                ping = (ping == P0) ? P1 : P0;
            }
        }
    }

    // ---- decode ----
    {
        int total = Bn * 184320;
        ssd_decode<<<(total + 255) / 256, 256, 0, stream>>>(
            LOC[0], LOC[1], LOC[2], LOC[3],
            CONF[0], CONF[1], CONF[2], CONF[3],
            prior, (float*)d_out, total);
    }
}
